// GNN_12000138625191
// MI455X (gfx1250) — matrix operand order repaired and checked by reading, not compiled, not run on hardware
//
#include <hip/hip_runtime.h>
#include <cstdint>
#include <cstddef>

#define Nn 250000
#define Ee 250000
#define Dd 128
#define Hh 256
#define HOPS 3
#define NTILES (Nn / 16)   // 15625
#define ETILES (Ee / 16)   // 15625

#define PBLK 512           // persistent blocks per big kernel
#define EW 8               // waves/block for 384-input kernels: LDS = 320KB exactly
#define ETPB (EW * 32)     // 256
#define VW 8               // waves/block for MLP_V
#define VTPB (VW * 32)     // 256

typedef __bf16 bf16_t;
typedef bf16_t v16bf __attribute__((ext_vector_type(16)));
typedef float  v8f   __attribute__((ext_vector_type(8)));

// ---------------------------------------------------------------------------
// WMMA helpers (CDNA5 wave32, v_wmma_f32_16x16x32_bf16)
// ---------------------------------------------------------------------------
static __device__ __forceinline__ v8f wmma_bf16(v16bf a, v16bf b, v8f c) {
  return __builtin_amdgcn_wmma_f32_16x16x32_bf16(false, a, false, b, (short)0, c, false, false);
}

// A fragment (16x32 bf16): lane l (m=l&15, half=l>>4):
//   elems 0..7  = K (kt*32 + half*8 + 0..7)
//   elems 8..15 = K (kt*32 + 16 + half*8 + 0..7)
static __device__ __forceinline__ v16bf load_a_bf(const bf16_t* rowkt, int lane) {
  const int half = (lane >> 4) & 1;
  union { v16bf v; uint4 q[2]; } u;
  u.q[0] = *(const uint4*)(rowkt + half * 8);
  u.q[1] = *(const uint4*)(rowkt + 16 + half * 8);
  return u.v;
}

static __device__ __forceinline__ v16bf load_a_f32(const float* rowkt, int lane) {
  const int half = (lane >> 4) & 1;
  const float* p0 = rowkt + half * 8;
  const float* p1 = rowkt + 16 + half * 8;
  v16bf r;
#pragma unroll
  for (int j = 0; j < 8; ++j) r[j] = (bf16_t)p0[j];
#pragma unroll
  for (int j = 0; j < 8; ++j) r[8 + j] = (bf16_t)p1[j];
  return r;
}

// Fused scatter-mean finalize: bf16( sum*scale + mask*token ) while building
// the A fragment.
static __device__ __forceinline__ v16bf load_a_scaled(const float* rowkt, float scale,
                                                      float mask, const float* tokkt,
                                                      int lane) {
  const int half = (lane >> 4) & 1;
  const float* p0 = rowkt + half * 8;
  const float* p1 = rowkt + 16 + half * 8;
  const float* t0 = tokkt + half * 8;
  const float* t1 = tokkt + 16 + half * 8;
  v16bf r;
#pragma unroll
  for (int j = 0; j < 8; ++j) r[j] = (bf16_t)(p0[j] * scale + mask * t0[j]);
#pragma unroll
  for (int j = 0; j < 8; ++j) r[8 + j] = (bf16_t)(p1[j] * scale + mask * t1[j]);
  return r;
}

// B fragments pre-packed: fragment fi = 32 lanes x 16 bf16 contiguous (1 KB).
static __device__ __forceinline__ v16bf load_b_frag(const bf16_t* packed, int fi, int lane) {
  const bf16_t* p = packed + ((size_t)fi * 32 + lane) * 16;
  union { v16bf v; uint4 q[2]; } u;
  u.q[0] = *(const uint4*)(p);
  u.q[1] = *(const uint4*)(p + 8);
  return u.v;
}

// Layer 1: KT k-tiles x 16 n-tiles. B frags chunked x4 (software pipelining:
// 4 ds-load pairs clause, WMMA i waits dscnt<=3-i; next chunk loads issue
// under current chunk WMMAs). A for kt+1 prefetched across the 16-WMMA chain.
template <int KT, typename F>
static __device__ __forceinline__ void layer1_run(v8f* acc1, const bf16_t* w1s,
                                                  int lane, F loadA) {
  v16bf a_cur = loadA(0);
#pragma unroll
  for (int kt = 0; kt < KT; ++kt) {
    v16bf a_nxt = (kt + 1 < KT) ? loadA(kt + 1) : a_cur;
#pragma unroll
    for (int n0 = 0; n0 < 16; n0 += 4) {
      v16bf b0 = load_b_frag(w1s, (n0 + 0) * KT + kt, lane);
      v16bf b1 = load_b_frag(w1s, (n0 + 1) * KT + kt, lane);
      v16bf b2 = load_b_frag(w1s, (n0 + 2) * KT + kt, lane);
      v16bf b3 = load_b_frag(w1s, (n0 + 3) * KT + kt, lane);
      acc1[n0 + 0] = wmma_bf16(a_cur, b0, acc1[n0 + 0]);
      acc1[n0 + 1] = wmma_bf16(a_cur, b1, acc1[n0 + 1]);
      acc1[n0 + 2] = wmma_bf16(a_cur, b2, acc1[n0 + 2]);
      acc1[n0 + 3] = wmma_bf16(a_cur, b3, acc1[n0 + 3]);
    }
    a_cur = a_nxt;
  }
}

// C/D layout: elem r on lane l -> (M = r + 8*(l>>4), N = l&15)
static __device__ __forceinline__ void store_l1(bf16_t* slice, const v8f* acc1,
                                                const float* b1, int lane) {
  const int half = (lane >> 4) & 1, nlo = lane & 15;
#pragma unroll
  for (int nt = 0; nt < 16; ++nt) {
    const float bias = b1[nt * 16 + nlo];
#pragma unroll
    for (int r = 0; r < 8; ++r) {
      float v = acc1[nt][r] + bias;
      slice[(r + 8 * half) * Hh + nt * 16 + nlo] = (bf16_t)fmaxf(v, 0.f);
    }
  }
}

// Layer 2: 8 k-tiles x 8 n-tiles; same chunked-B + prefetched-A pipelining.
static __device__ __forceinline__ void layer2_run(const bf16_t* slice, const bf16_t* w2s,
                                                  v8f* acc2, int lane) {
  const bf16_t* arow = slice + (lane & 15) * Hh;
  v16bf a_cur = load_a_bf(arow, lane);
#pragma unroll
  for (int kt = 0; kt < 8; ++kt) {
    v16bf a_nxt = (kt + 1 < 8) ? load_a_bf(arow + (kt + 1) * 32, lane) : a_cur;
#pragma unroll
    for (int n0 = 0; n0 < 8; n0 += 4) {
      v16bf b0 = load_b_frag(w2s, (n0 + 0) * 8 + kt, lane);
      v16bf b1 = load_b_frag(w2s, (n0 + 1) * 8 + kt, lane);
      v16bf b2 = load_b_frag(w2s, (n0 + 2) * 8 + kt, lane);
      v16bf b3 = load_b_frag(w2s, (n0 + 3) * 8 + kt, lane);
      acc2[n0 + 0] = wmma_bf16(a_cur, b0, acc2[n0 + 0]);
      acc2[n0 + 1] = wmma_bf16(a_cur, b1, acc2[n0 + 1]);
      acc2[n0 + 2] = wmma_bf16(a_cur, b2, acc2[n0 + 2]);
      acc2[n0 + 3] = wmma_bf16(a_cur, b3, acc2[n0 + 3]);
    }
    a_cur = a_nxt;
  }
}

// Cooperative global->LDS staging of packed weights (one-time per block).
static __device__ __forceinline__ void stage(bf16_t* dst, const bf16_t* src, int nelem) {
  uint4* d = (uint4*)dst;
  const uint4* s = (const uint4*)src;
  const int n4 = nelem >> 3;
  for (int i = threadIdx.x; i < n4; i += blockDim.x) d[i] = s[i];
}

// ---------------------------------------------------------------------------
// Weight repack: fp32 [K,N] row-major -> bf16 B-fragment order.
// ---------------------------------------------------------------------------
__global__ void k_pack(const float* __restrict__ src, bf16_t* __restrict__ dst,
                       int K, int Ncols, int total) {
  int tid = blockIdx.x * blockDim.x + threadIdx.x;
  if (tid >= total) return;
  const int j = tid & 15;
  const int lane = (tid >> 4) & 31;
  const int fi = tid >> 9;
  const int KT = K >> 5;
  const int kt = fi % KT;
  const int nt = fi / KT;
  const int n = nt * 16 + (lane & 15);
  const int k = kt * 32 + ((j >> 3) << 4) + ((lane >> 4) << 3) + (j & 7);
  dst[tid] = (bf16_t)src[(size_t)k * Ncols + n];
}

// edge_in = MLP_E(1), edge_out = MLP_E(0)
__global__ void k_edgevec(const float* __restrict__ Ew1, const float* __restrict__ Eb1,
                          const float* __restrict__ Ew2, const float* __restrict__ Eb2,
                          bf16_t* __restrict__ ein, bf16_t* __restrict__ eout) {
  __shared__ float h1[Hh], h0[Hh];
  const int t = threadIdx.x;  // blockDim = 256
  h1[t] = fmaxf(Ew1[t] + Eb1[t], 0.f);
  h0[t] = fmaxf(Eb1[t], 0.f);
  __syncthreads();
  if (t < Dd) {
    float ai = Eb2[t], ao = Eb2[t];
    for (int h = 0; h < Hh; ++h) {
      const float w = Ew2[h * Dd + t];
      ai += h1[h] * w;
      ao += h0[h] * w;
    }
    ein[t]  = (bf16_t)fmaxf(ai, 0.f);
    eout[t] = (bf16_t)fmaxf(ao, 0.f);
  }
}

__global__ void k_count(const int* __restrict__ sidx, const int* __restrict__ pidx,
                        unsigned* __restrict__ cnt_p, unsigned* __restrict__ cnt_c) {
  int i = blockIdx.x * blockDim.x + threadIdx.x;
  if (i < Ee) {
    atomicAdd(&cnt_p[sidx[i]], 1u);
    atomicAdd(&cnt_c[pidx[i]], 1u);
  }
}

// ---------------------------------------------------------------------------
// hidden = MLP_V(batch_token). Persistent; weights resident in LDS (192KB).
// ---------------------------------------------------------------------------
__global__ __launch_bounds__(VTPB) void k_mlp_v(
    const float* __restrict__ bt,
    const bf16_t* __restrict__ w1g, const float* __restrict__ b1,
    const bf16_t* __restrict__ w2g, const float* __restrict__ b2,
    float* __restrict__ hid_f, bf16_t* __restrict__ hid_bf) {
  __shared__ __align__(16) bf16_t w1s[128 * 256];
  __shared__ __align__(16) bf16_t w2s[256 * 128];
  __shared__ __align__(16) bf16_t act[VW * 16 * Hh];
  stage(w1s, w1g, 128 * 256);
  stage(w2s, w2g, 256 * 128);
  __syncthreads();

  const int wave = threadIdx.x >> 5, lane = threadIdx.x & 31;
  bf16_t* slice = act + wave * 16 * Hh;
  const int m = lane & 15, half = (lane >> 4) & 1;
  const v8f VZ = {0.f, 0.f, 0.f, 0.f, 0.f, 0.f, 0.f, 0.f};

  for (int tile = blockIdx.x * VW + wave; tile < NTILES; tile += gridDim.x * VW) {
    const int tb = tile * 16;
    const float* arow = bt + (size_t)(tb + m) * Dd;
    v8f acc1[16];
#pragma unroll
    for (int i = 0; i < 16; ++i) acc1[i] = VZ;
    layer1_run<4>(acc1, w1s, lane,
                  [&](int kt) { return load_a_f32(arow + kt * 32, lane); });
    store_l1(slice, acc1, b1, lane);

    v8f acc2[8];
#pragma unroll
    for (int i = 0; i < 8; ++i) acc2[i] = VZ;
    layer2_run(slice, w2s, acc2, lane);

#pragma unroll
    for (int nt = 0; nt < 8; ++nt) {
      const float bias = b2[nt * 16 + m];
#pragma unroll
      for (int r = 0; r < 8; ++r) {
        const float v = fmaxf(acc2[nt][r] + bias, 0.f);
        const size_t o = (size_t)(tb + r + 8 * half) * Dd + nt * 16 + m;
        hid_f[o] = v;
        hid_bf[o] = (bf16_t)v;
      }
    }
  }
}

// ---------------------------------------------------------------------------
// One edge MLP pass ([g0 | g1 | evec] 384 -> 256 -> 128, scatter-add).
// Persistent; LDS: 192KB w1 + 64KB w2 + 64KB act = 320KB exactly (8 waves).
// ---------------------------------------------------------------------------
__global__ __launch_bounds__(ETPB) void k_edge_one(
    const bf16_t* __restrict__ hid_bf,
    const int* __restrict__ g0_idx, const int* __restrict__ g1_idx,
    const bf16_t* __restrict__ evec, const int* __restrict__ scat_idx,
    const bf16_t* __restrict__ w1g, const float* __restrict__ b1,
    const bf16_t* __restrict__ w2g, const float* __restrict__ b2,
    float* __restrict__ out_sum) {
  __shared__ __align__(16) bf16_t w1s[384 * 256];
  __shared__ __align__(16) bf16_t w2s[256 * 128];
  __shared__ __align__(16) bf16_t act[EW * 16 * Hh];
  stage(w1s, w1g, 384 * 256);
  stage(w2s, w2g, 256 * 128);
  __syncthreads();

  const int wave = threadIdx.x >> 5, lane = threadIdx.x & 31;
  bf16_t* slice = act + wave * 16 * Hh;
  const int m = lane & 15, half = (lane >> 4) & 1;
  const v8f VZ = {0.f, 0.f, 0.f, 0.f, 0.f, 0.f, 0.f, 0.f};

  for (int tile = blockIdx.x * EW + wave; tile < ETILES; tile += gridDim.x * EW) {
    const int tb = tile * 16;
    const bf16_t* a0 = hid_bf + (size_t)g0_idx[tb + m] * Dd;
    const bf16_t* a1 = hid_bf + (size_t)g1_idx[tb + m] * Dd;

    v8f acc1[16];
#pragma unroll
    for (int i = 0; i < 16; ++i) acc1[i] = VZ;
    layer1_run<12>(acc1, w1s, lane, [&](int kt) {
      const bf16_t* base = (kt < 4) ? a0 : (kt < 8) ? a1 : evec;
      return load_a_bf(base + (kt & 3) * 32, lane);
    });
    store_l1(slice, acc1, b1, lane);

    v8f acc2[8];
#pragma unroll
    for (int i = 0; i < 8; ++i) acc2[i] = VZ;
    layer2_run(slice, w2s, acc2, lane);

    int rows[8];
#pragma unroll
    for (int r = 0; r < 8; ++r) rows[r] = scat_idx[tb + r + 8 * half];
#pragma unroll
    for (int nt = 0; nt < 8; ++nt) {
      const float bias = b2[nt * 16 + m];
#pragma unroll
      for (int r = 0; r < 8; ++r) {
        const float v = fmaxf(acc2[nt][r] + bias, 0.f);
        unsafeAtomicAdd(&out_sum[(size_t)rows[r] * Dd + nt * 16 + m], v);
      }
    }
  }
}

// ---------------------------------------------------------------------------
// hidden += MLP_A([hidden | mean(S_p)+root*start | mean(S_c)+leaf*end])
// Scatter-mean finalize fused into the A-fragment loads.
// Persistent; LDS = 320KB exactly (8 waves).
// ---------------------------------------------------------------------------
__global__ __launch_bounds__(ETPB) void k_node(
    const float* __restrict__ sp_sum, const float* __restrict__ sc_sum,
    const unsigned* __restrict__ cnt_p, const unsigned* __restrict__ cnt_c,
    const float* __restrict__ root_mask, const float* __restrict__ leaf_mask,
    const float* __restrict__ start_tok, const float* __restrict__ end_tok,
    const bf16_t* __restrict__ w1g, const float* __restrict__ b1,
    const bf16_t* __restrict__ w2g, const float* __restrict__ b2,
    float* hid_f, bf16_t* hid_bf) {
  __shared__ __align__(16) bf16_t w1s[384 * 256];
  __shared__ __align__(16) bf16_t w2s[256 * 128];
  __shared__ __align__(16) bf16_t act[EW * 16 * Hh];
  stage(w1s, w1g, 384 * 256);
  stage(w2s, w2g, 256 * 128);
  __syncthreads();

  const int wave = threadIdx.x >> 5, lane = threadIdx.x & 31;
  bf16_t* slice = act + wave * 16 * Hh;
  const int m = lane & 15, half = (lane >> 4) & 1;
  const v8f VZ = {0.f, 0.f, 0.f, 0.f, 0.f, 0.f, 0.f, 0.f};

  for (int tile = blockIdx.x * EW + wave; tile < NTILES; tile += gridDim.x * EW) {
    const int tb = tile * 16;
    const int row = tb + m;
    const bf16_t* h0 = hid_bf + (size_t)row * Dd;
    const float* sp = sp_sum + (size_t)row * Dd;
    const float* sc = sc_sum + (size_t)row * Dd;
    const float invp = 1.f / fmaxf((float)cnt_p[row], 1.f);
    const float invc = 1.f / fmaxf((float)cnt_c[row], 1.f);
    const float rm = root_mask[row];
    const float lm = leaf_mask[row];

    v8f acc1[16];
#pragma unroll
    for (int i = 0; i < 16; ++i) acc1[i] = VZ;
    layer1_run<12>(acc1, w1s, lane, [&](int kt) {
      const int ko = (kt & 3) * 32;
      if (kt < 4) return load_a_bf(h0 + ko, lane);
      if (kt < 8) return load_a_scaled(sp + ko, invp, rm, start_tok + ko, lane);
      return load_a_scaled(sc + ko, invc, lm, end_tok + ko, lane);
    });
    store_l1(slice, acc1, b1, lane);

    v8f acc2[8];
#pragma unroll
    for (int i = 0; i < 8; ++i) acc2[i] = VZ;
    layer2_run(slice, w2s, acc2, lane);

#pragma unroll
    for (int nt = 0; nt < 8; ++nt) {
      const float bias = b2[nt * 16 + m];
#pragma unroll
      for (int r = 0; r < 8; ++r) {
        const size_t o = (size_t)(tb + r + 8 * half) * Dd + nt * 16 + m;
        const float v = hid_f[o] + fmaxf(acc2[nt][r] + bias, 0.f);
        hid_f[o] = v;
        hid_bf[o] = (bf16_t)v;
      }
    }
  }
}

// ---------------------------------------------------------------------------
// Host launch
// ---------------------------------------------------------------------------
extern "C" void kernel_launch(void* const* d_in, const int* in_sizes, int n_in,
                              void* d_out, int out_size, void* d_ws, size_t ws_size,
                              hipStream_t stream) {
  const float* bt        = (const float*)d_in[0];
  const int*   sidx      = (const int*)d_in[1];
  const int*   pidx      = (const int*)d_in[2];
  const float* root_mask = (const float*)d_in[3];
  const float* leaf_mask = (const float*)d_in[4];
  const float* start_tok = (const float*)d_in[5];
  const float* end_tok   = (const float*)d_in[6];
  const float* Vw1 = (const float*)d_in[7],  *Vb1 = (const float*)d_in[8];
  const float* Vw2 = (const float*)d_in[9],  *Vb2 = (const float*)d_in[10];
  const float* Ew1 = (const float*)d_in[11], *Eb1 = (const float*)d_in[12];
  const float* Ew2 = (const float*)d_in[13], *Eb2 = (const float*)d_in[14];
  const float* Pw1 = (const float*)d_in[15], *Pb1 = (const float*)d_in[16];
  const float* Pw2 = (const float*)d_in[17], *Pb2 = (const float*)d_in[18];
  const float* Cw1 = (const float*)d_in[19], *Cb1 = (const float*)d_in[20];
  const float* Cw2 = (const float*)d_in[21], *Cb2 = (const float*)d_in[22];
  const float* Aw1 = (const float*)d_in[23], *Ab1 = (const float*)d_in[24];
  const float* Aw2 = (const float*)d_in[25], *Ab2 = (const float*)d_in[26];
  float* hid_f = (float*)d_out;

  char* w = (char*)d_ws;
  auto bump = [&](size_t bytes) -> char* {
    char* p = w;
    w += (bytes + 255) & ~(size_t)255;
    return p;
  };
  float*    sp_sum = (float*)bump((size_t)Nn * Dd * 4);
  float*    sc_sum = (float*)bump((size_t)Nn * Dd * 4);
  bf16_t*   hid_bf = (bf16_t*)bump((size_t)Nn * Dd * 2);
  unsigned* cnt_p  = (unsigned*)bump((size_t)Nn * 4);
  unsigned* cnt_c  = (unsigned*)bump((size_t)Nn * 4);
  bf16_t* vw1p = (bf16_t*)bump(128 * 256 * 2);
  bf16_t* vw2p = (bf16_t*)bump(256 * 128 * 2);
  bf16_t* pw1p = (bf16_t*)bump(384 * 256 * 2);
  bf16_t* pw2p = (bf16_t*)bump(256 * 128 * 2);
  bf16_t* cw1p = (bf16_t*)bump(384 * 256 * 2);
  bf16_t* cw2p = (bf16_t*)bump(256 * 128 * 2);
  bf16_t* aw1p = (bf16_t*)bump(384 * 256 * 2);
  bf16_t* aw2p = (bf16_t*)bump(256 * 128 * 2);
  bf16_t* ein  = (bf16_t*)bump(Dd * 2);
  bf16_t* eout = (bf16_t*)bump(Dd * 2);

  auto pack = [&](const float* src, bf16_t* dst, int K, int Ncols) {
    const int total = K * Ncols;
    k_pack<<<(total + 255) / 256, 256, 0, stream>>>(src, dst, K, Ncols, total);
  };
  pack(Vw1, vw1p, 128, 256);
  pack(Vw2, vw2p, 256, 128);
  pack(Pw1, pw1p, 384, 256);
  pack(Pw2, pw2p, 256, 128);
  pack(Cw1, cw1p, 384, 256);
  pack(Cw2, cw2p, 256, 128);
  pack(Aw1, aw1p, 384, 256);
  pack(Aw2, aw2p, 256, 128);

  k_edgevec<<<1, 256, 0, stream>>>(Ew1, Eb1, Ew2, Eb2, ein, eout);

  hipMemsetAsync(cnt_p, 0, (size_t)Nn * 4, stream);
  hipMemsetAsync(cnt_c, 0, (size_t)Nn * 4, stream);
  k_count<<<(Ee + 255) / 256, 256, 0, stream>>>(sidx, pidx, cnt_p, cnt_c);

  k_mlp_v<<<PBLK, VTPB, 0, stream>>>(bt, vw1p, Vb1, vw2p, Vb2, hid_f, hid_bf);

  for (int hop = 0; hop < HOPS; ++hop) {
    hipMemsetAsync(sp_sum, 0, (size_t)Nn * Dd * 4, stream);
    hipMemsetAsync(sc_sum, 0, (size_t)Nn * Dd * 4, stream);
    // S_p = MLP_P([parent | self | edge_out]) scattered by self_idx
    k_edge_one<<<PBLK, ETPB, 0, stream>>>(hid_bf, pidx, sidx, eout, sidx,
                                          pw1p, Pb1, pw2p, Pb2, sp_sum);
    // S_c = MLP_C([self | parent | edge_in]) scattered by parent_idx
    k_edge_one<<<PBLK, ETPB, 0, stream>>>(hid_bf, sidx, pidx, ein, pidx,
                                          cw1p, Cb1, cw2p, Cb2, sc_sum);
    k_node<<<PBLK, ETPB, 0, stream>>>(sp_sum, sc_sum, cnt_p, cnt_c,
                                      root_mask, leaf_mask, start_tok, end_tok,
                                      aw1p, Ab1, aw2p, Ab2, hid_f, hid_bf);
  }
}
